// VQVAE_54511724921600
// MI455X (gfx1250) — matrix operand order repaired and checked by reading, not compiled, not run on hardware
//
#include <hip/hip_runtime.h>
#include <hip/hip_bf16.h>

typedef __attribute__((ext_vector_type(16))) _Float16 v16h;
typedef __attribute__((ext_vector_type(8)))  float    v8f;

__device__ __forceinline__ float lrelu_f(float v) { return v > 0.f ? v : 0.01f * v; }

// ---------------------------------------------------------------------------
// CDNA5 async global->LDS staging (ASYNCcnt path).
// Generic __shared__ pointers: low 32 bits == LDS byte address (ISA flat
// aperture rule).
// ---------------------------------------------------------------------------
__device__ __forceinline__ void async_stage_b128(unsigned lds_byte, const void* gaddr)
{
    asm volatile("global_load_async_to_lds_b128 %0, %1, off"
                 :: "v"(lds_byte), "v"(gaddr) : "memory");
}
__device__ __forceinline__ void async_wait_all()
{
    asm volatile("s_wait_asynccnt 0" ::: "memory");
}

// Cooperatively stage n_f16 half-words from packed global into LDS.
__device__ __forceinline__ void stage_f16(const _Float16* __restrict__ src,
                                          _Float16* dst_lds, int n_f16)
{
    const unsigned base = (unsigned)(unsigned long long)(uintptr_t)dst_lds;
    const int chunks = n_f16 >> 3;                     // 16B per b128
    for (int c = threadIdx.x; c < chunks; c += blockDim.x) {
        async_stage_b128(base + (unsigned)(c << 4),
                         (const void*)((const char*)src + ((long)c << 4)));
    }
    async_wait_all();
    __syncthreads();
}

// ---------------------------------------------------------------------------
// Pack f32 weights (OIHW) / codebook rows into the WMMA B-fragment f16 layout:
//   linear f16 index i = ((tn*KCH + kc)*32 + lane)*16 + j
//   lane half g = lane>>4 holds K block [16g..16g+15]; column co = tn*16 + lane%16
//   src element = w[co*K + (kc*32 + g*16 + j)]   (K-order == CIN*KH*KW flattening)
// ---------------------------------------------------------------------------
__global__ __launch_bounds__(256)
void pack_wgt_k(const float* __restrict__ w, _Float16* __restrict__ wp,
                int COUT, int K)
{
    const int i = blockIdx.x * 256 + threadIdx.x;
    if (i >= COUT * K) return;
    const int j    = i & 15;
    const int lane = (i >> 4) & 31;
    const int t    = i >> 9;               // tn*KCH + kc
    const int KCH  = K >> 5;
    const int tn = t / KCH, kc = t % KCH;
    const int g = lane >> 4, half = lane & 15;
    const int co = tn * 16 + half;
    const int kb = kc * 32 + g * 16 + j;
    wp[i] = (_Float16)w[(long)co * K + kb];
}

// ---------------------------------------------------------------------------
// Implicit-GEMM conv on the WMMA pipe. 16x16 output tile per wave, 8 waves per
// block all sharing one N-tile, whose packed f16 weight slice (16*K halves) is
// async-staged into LDS once and fragment-read with ds_load_b128.
//   A fragment (16x32 f16): lane half g holds K block [16g..16g+15]
//   C/D: lane n = L%16, VGPR r -> row m = r + 8*(L/16)
// Optional fused 2x align-corners bilinear upsample of the input, bias,
// residual add, leaky/relu epilogue.
// ---------------------------------------------------------------------------
template<int CIN, int COUT, int KH, int KW, int STRIDE, int PAD,
         int HIN, int WIN, int HOUT, int WOUT, bool UPS>
__global__ __launch_bounds__(256)
void wmma_conv_k(const float* __restrict__ x, const _Float16* __restrict__ wp,
                 const float* __restrict__ bias, const float* __restrict__ resid,
                 float* __restrict__ y, int act)
{
    constexpr int KK  = KH * KW;
    constexpr int K   = CIN * KK;
    static_assert(K % 32 == 0, "K must be a multiple of 32");
    constexpr int KCH = K / 32;
    constexpr int MT  = (128 * HOUT * WOUT) / 16;      // pixel tiles
    constexpr int HS  = UPS ? HIN / 2 : HIN;           // stored input dims
    constexpr int WS  = UPS ? WIN / 2 : WIN;

    __shared__ _Float16 swgt[16 * K];                  // one N-slice, <= 18KB
    static_assert(16 * K * 2 <= 64 * 1024, "LDS slice too big");

    const int lane = threadIdx.x & 31;
    const int wid  = threadIdx.x >> 5;
    const int tile = blockIdx.x * (blockDim.x >> 5) + wid;
    const int tn   = tile / MT;                        // uniform across block
    const int tm   = tile % MT;
    const int m0   = tm * 16;
    const int g    = lane >> 4;
    const int half = lane & 15;

    stage_f16(wp + (long)tn * (16 * K), swgt, 16 * K);

    // A-row pixel for this lane (row m = half)
    const int pa = m0 + half;
    const int wa = pa % WOUT;
    const int ha = (pa / WOUT) % HOUT;
    const int na = pa / (WOUT * HOUT);
    const int co = tn * 16 + half;                     // store column

    auto fetch = [&](int c, int h, int w) -> float {
        if (h < 0 || h >= HIN || w < 0 || w >= WIN) return 0.0f;
        const float* base = x + ((long)na * CIN + c) * (long)(HS * WS);
        if constexpr (UPS) {
            const float sy = h * ((float)(HS - 1) / (float)(HIN - 1));
            const float sx = w * ((float)(WS - 1) / (float)(WIN - 1));
            const int y0 = (int)sy; const int y1 = (y0 + 1 < HS) ? y0 + 1 : HS - 1;
            const int x0 = (int)sx; const int x1 = (x0 + 1 < WS) ? x0 + 1 : WS - 1;
            const float fy = sy - (float)y0, fx = sx - (float)x0;
            const float v00 = base[y0 * WS + x0], v01 = base[y0 * WS + x1];
            const float v10 = base[y1 * WS + x0], v11 = base[y1 * WS + x1];
            const float t0 = v00 + (v01 - v00) * fx;
            const float t1 = v10 + (v11 - v10) * fx;
            return t0 + (t1 - t0) * fy;
        } else {
            return base[h * WS + w];
        }
    };

    const v16h* wv = (const v16h*)swgt;
    v8f acc = {};
    for (int kc = 0; kc < KCH; ++kc) {
        v16h a;
#pragma unroll
        for (int j = 0; j < 16; ++j) {
            const int ka = kc * 32 + g * 16 + j;
            const int ci = ka / KK, rs = ka % KK, r = rs / KW, s = rs % KW;
            const int ih = ha * STRIDE - PAD + r;
            const int iw = wa * STRIDE - PAD + s;
            a[j] = (_Float16)fetch(ci, ih, iw);
        }
        const v16h b = wv[kc * 32 + lane];             // ds_load_b128 x2
        acc = __builtin_amdgcn_wmma_f32_16x16x32_f16(false, a, false, b,
                                                     (short)0, acc, false, false);
    }

    const float bv = bias ? bias[co] : 0.0f;
#pragma unroll
    for (int r2 = 0; r2 < 8; ++r2) {
        const int m = r2 + 8 * g;
        const int p = m0 + m;
        const int w = p % WOUT;
        const int h = (p / WOUT) % HOUT;
        const int n = p / (WOUT * HOUT);
        const long oi = (((long)n * COUT + co) * HOUT + h) * WOUT + w;
        float v = acc[r2] + bv;
        if (resid) v += resid[oi];
        if (act == 1)      v = lrelu_f(v);
        else if (act == 2) v = v > 0.f ? v : 0.f;
        y[oi] = v;
    }
}

// ---------------------------------------------------------------------------
// Encoder stem: Cin=1 4x4 s2 p1 conv + lrelu (K=16, not worth the matrix pipe)
// ---------------------------------------------------------------------------
__global__ __launch_bounds__(256)
void conv1_k(const float* __restrict__ x, const float* __restrict__ w,
             const float* __restrict__ b, float* __restrict__ y)
{
    const long i = (long)blockIdx.x * 256 + threadIdx.x;   // 16777216 total
    const int wo = (int)(i & 63);
    const int ho = (int)((i >> 6) & 63);
    const int co = (int)((i >> 12) & 31);
    const int n  = (int)(i >> 17);
    const float* xb = x + (long)n * 16384;
    float s = b[co];
#pragma unroll
    for (int r = 0; r < 4; ++r) {
        const int ih = ho * 2 - 1 + r;
        if (ih < 0 || ih >= 128) continue;
#pragma unroll
        for (int t = 0; t < 4; ++t) {
            const int iw = wo * 2 - 1 + t;
            if (iw < 0 || iw >= 128) continue;
            s += xb[ih * 128 + iw] * w[co * 16 + r * 4 + t];
        }
    }
    y[i] = lrelu_f(s);
}

// ---------------------------------------------------------------------------
// Codebook squared norms (512 x 64)
// ---------------------------------------------------------------------------
__global__ __launch_bounds__(256)
void cnorm_k(const float* __restrict__ cb, float* __restrict__ cn)
{
    const int n = blockIdx.x * 256 + threadIdx.x;
    if (n >= 512) return;
    float s = 0.f;
    for (int d = 0; d < 64; ++d) { const float v = cb[n * 64 + d]; s += v * v; }
    cn[n] = s;
}

// ---------------------------------------------------------------------------
// VQ nearest-codeword search on the WMMA pipe. The whole packed f16 codebook
// (64KB, B-fragment layout) is async-staged into LDS once per block and reused
// by all 8 pixel-tile waves across all 32 codeword tiles.
// argmin over (||c||^2 - 2 x.c); cross-lane xor-shuffle argmin reduction.
// ---------------------------------------------------------------------------
__global__ __launch_bounds__(256)
void vq_argmin_k(const float* __restrict__ lat, const _Float16* __restrict__ cbp,
                 const float* __restrict__ cnorm, int* __restrict__ inds)
{
    __shared__ _Float16 cbs[512 * 64];                 // 64KB packed codebook

    const int lane = threadIdx.x & 31;
    const int wid  = threadIdx.x >> 5;
    const int tm   = blockIdx.x * (blockDim.x >> 5) + wid;
    const int m0   = tm * 16;
    const int g    = lane >> 4;
    const int half = lane & 15;

    stage_f16(cbp, cbs, 512 * 64);

    const int pa = m0 + half;
    const int wa = pa & 31, ha = (pa >> 5) & 31, na = pa >> 10;
    const float* latb = lat + (long)na * 65536 + ha * 32 + wa;  // + d*1024

    float minv[8]; int mini[8];
#pragma unroll
    for (int r = 0; r < 8; ++r) { minv[r] = 3.0e38f; mini[r] = 0; }

    const v16h* cbv = (const v16h*)cbs;
    for (int tn = 0; tn < 32; ++tn) {
        const int n = tn * 16 + half;
        v8f acc = {};
#pragma unroll
        for (int kc = 0; kc < 2; ++kc) {
            v16h a;
#pragma unroll
            for (int j = 0; j < 16; ++j) {
                const int ka = kc * 32 + g * 16 + j;
                a[j] = (_Float16)latb[(long)ka * 1024];
            }
            const v16h b = cbv[(tn * 2 + kc) * 32 + lane];
            acc = __builtin_amdgcn_wmma_f32_16x16x32_f16(false, a, false, b,
                                                         (short)0, acc, false, false);
        }
        const float cn = cnorm[n];
#pragma unroll
        for (int r = 0; r < 8; ++r) {
            const float v = cn - 2.0f * acc[r];
            if (v < minv[r] || (v == minv[r] && n < mini[r])) { minv[r] = v; mini[r] = n; }
        }
    }
#pragma unroll
    for (int off = 8; off >= 1; off >>= 1) {
#pragma unroll
        for (int r = 0; r < 8; ++r) {
            const float ov = __shfl_xor(minv[r], off, 32);
            const int   oi = __shfl_xor(mini[r], off, 32);
            if (ov < minv[r] || (ov == minv[r] && oi < mini[r])) { minv[r] = ov; mini[r] = oi; }
        }
    }
    if (half == 0) {
#pragma unroll
        for (int r = 0; r < 8; ++r) inds[m0 + r + 8 * g] = mini[r];
    }
}

// ---------------------------------------------------------------------------
// Gather quantized latents + per-block partial sums of (q - lat)^2.
// ---------------------------------------------------------------------------
__global__ __launch_bounds__(256)
void vq_gather_k(const float* __restrict__ lat, const float* __restrict__ cb,
                 const int* __restrict__ inds, float* __restrict__ q,
                 float* __restrict__ bsum)
{
    __shared__ float sm[256];
    const long i = (long)blockIdx.x * 256 + threadIdx.x;   // 8388608 total
    const int n  = (int)(i >> 16);
    const int d  = (int)((i >> 10) & 63);
    const int hw = (int)(i & 1023);
    const int p  = n * 1024 + hw;
    const float l  = lat[i];
    const float qq = cb[(long)inds[p] * 64 + d];
    q[i] = qq;
    const float dd = qq - l;
    sm[threadIdx.x] = dd * dd;
    __syncthreads();
    for (int s = 128; s > 0; s >>= 1) {
        if (threadIdx.x < s) sm[threadIdx.x] += sm[threadIdx.x + s];
        __syncthreads();
    }
    if (threadIdx.x == 0) bsum[blockIdx.x] = sm[0];
}

__global__ __launch_bounds__(256)
void vq_loss_reduce_k(const float* __restrict__ bsum, int nb,
                      float* __restrict__ out_losses)
{
    __shared__ float sm[256];
    float s = 0.f;
    for (int i = threadIdx.x; i < nb; i += 256) s += bsum[i];
    sm[threadIdx.x] = s;
    __syncthreads();
    for (int st = 128; st > 0; st >>= 1) {
        if (threadIdx.x < st) sm[threadIdx.x] += sm[threadIdx.x + st];
        __syncthreads();
    }
    if (threadIdx.x == 0) {
        const float mse = sm[0] / 8388608.0f;
        out_losses[0] = mse;            // embedding loss (forward)
        out_losses[1] = 0.25f * mse;    // BETA * commitment (forward)
    }
}

// ---------------------------------------------------------------------------
// Final conv: fused 2x align-corners upsample (from 64x64) + 3x3 32->1 + ReLU
// ---------------------------------------------------------------------------
__global__ __launch_bounds__(256)
void dec3_k(const float* __restrict__ d3, const float* __restrict__ w,
            const float* __restrict__ b, float* __restrict__ out)
{
    const long i = (long)blockIdx.x * 256 + threadIdx.x;   // 2097152 total
    const int wo = (int)(i & 127);
    const int ho = (int)((i >> 7) & 127);
    const int n  = (int)(i >> 14);
    float s = b[0];
    for (int ci = 0; ci < 32; ++ci) {
        const float* base = d3 + ((long)n * 32 + ci) * 4096;
#pragma unroll
        for (int r = 0; r < 3; ++r) {
            const int h = ho - 1 + r;
            if (h < 0 || h >= 128) continue;
            const float sy = h * (63.0f / 127.0f);
            const int y0 = (int)sy; const int y1 = (y0 + 1 < 64) ? y0 + 1 : 63;
            const float fy = sy - (float)y0;
#pragma unroll
            for (int t = 0; t < 3; ++t) {
                const int ww = wo - 1 + t;
                if (ww < 0 || ww >= 128) continue;
                const float sx = ww * (63.0f / 127.0f);
                const int x0 = (int)sx; const int x1 = (x0 + 1 < 64) ? x0 + 1 : 63;
                const float fx = sx - (float)x0;
                const float v00 = base[y0 * 64 + x0], v01 = base[y0 * 64 + x1];
                const float v10 = base[y1 * 64 + x0], v11 = base[y1 * 64 + x1];
                const float t0 = v00 + (v01 - v00) * fx;
                const float t1 = v10 + (v11 - v10) * fx;
                s += (t0 + (t1 - t0) * fy) * w[ci * 9 + r * 3 + t];
            }
        }
    }
    out[i] = s > 0.f ? s : 0.f;
}

// ---------------------------------------------------------------------------
extern "C" void kernel_launch(void* const* d_in, const int* in_sizes, int n_in,
                              void* d_out, int out_size, void* d_ws, size_t ws_size,
                              hipStream_t stream)
{
    (void)in_sizes; (void)n_in; (void)out_size; (void)ws_size;

    const float* inp  = (const float*)d_in[0];
    const float* ew1  = (const float*)d_in[1];  const float* eb1 = (const float*)d_in[2];
    const float* ew2  = (const float*)d_in[3];  const float* eb2 = (const float*)d_in[4];
    const float* ew3  = (const float*)d_in[5];  const float* eb3 = (const float*)d_in[6];
    const float* rew1 = (const float*)d_in[7];
    const float* rew2 = (const float*)d_in[8];
    const float* ew4  = (const float*)d_in[9];  const float* eb4 = (const float*)d_in[10];
    const float* cb   = (const float*)d_in[11];
    const float* dw1  = (const float*)d_in[12]; const float* db1 = (const float*)d_in[13];
    const float* rdw1 = (const float*)d_in[14];
    const float* rdw2 = (const float*)d_in[15];
    const float* dw2  = (const float*)d_in[16]; const float* db2 = (const float*)d_in[17];
    const float* dw3  = (const float*)d_in[18]; const float* db3 = (const float*)d_in[19];

    // ---- workspace layout (f32 region, then packed-f16 region) ----
    float* ws   = (float*)d_ws;
    float* buf0 = ws;                      // 16777216  [128,32,64,64]
    float* buf1 = buf0 + 16777216;         //  8388608  [128,64,32,32]
    float* buf2 = buf1 + 8388608;
    float* buf3 = buf2 + 8388608;
    float* cn   = buf3 + 8388608;          //      512
    float* bsum = cn + 512;                //    32768
    int*   inds = (int*)(bsum + 32768);    //   131072
    _Float16* hp = (_Float16*)(ws + 42107392);  // 16B-aligned f16 pack region
    _Float16* ew2p  = hp;                  // 64*512  = 32768
    _Float16* ew3p  = hp + 32768;          // 64*576  = 36864
    _Float16* rew1p = hp + 69632;          // 36864
    _Float16* rew2p = hp + 106496;         // 64*64   = 4096
    _Float16* ew4p  = hp + 110592;         // 4096
    _Float16* dw1p  = hp + 114688;         // 36864
    _Float16* rdw1p = hp + 151552;         // 36864
    _Float16* rdw2p = hp + 188416;         // 4096
    _Float16* dw2p  = hp + 192512;         // 32*576  = 18432
    _Float16* cbp   = hp + 210944;         // 512*64  = 32768
    float* recon  = (float*)d_out;         //  2097152
    float* losses = recon + 2097152;       //        2

    // ---- pre-pack all B operands into WMMA fragment layout (f16) ----
    pack_wgt_k<<<128, 256, 0, stream>>>(ew2,  ew2p,  64, 512);
    pack_wgt_k<<<144, 256, 0, stream>>>(ew3,  ew3p,  64, 576);
    pack_wgt_k<<<144, 256, 0, stream>>>(rew1, rew1p, 64, 576);
    pack_wgt_k<<< 16, 256, 0, stream>>>(rew2, rew2p, 64, 64);
    pack_wgt_k<<< 16, 256, 0, stream>>>(ew4,  ew4p,  64, 64);
    pack_wgt_k<<<144, 256, 0, stream>>>(dw1,  dw1p,  64, 576);
    pack_wgt_k<<<144, 256, 0, stream>>>(rdw1, rdw1p, 64, 576);
    pack_wgt_k<<< 16, 256, 0, stream>>>(rdw2, rdw2p, 64, 64);
    pack_wgt_k<<< 72, 256, 0, stream>>>(dw2,  dw2p,  32, 576);
    pack_wgt_k<<<128, 256, 0, stream>>>(cb,   cbp,   512, 64);
    cnorm_k<<<2, 256, 0, stream>>>(cb, cn);

    // ---- encoder ----
    conv1_k<<<65536, 256, 0, stream>>>(inp, ew1, eb1, buf0);
    wmma_conv_k<32,64,4,4,2,1, 64,64, 32,32,false><<<4096,256,0,stream>>>(buf0, ew2p,  eb2, nullptr, buf1, 1);
    wmma_conv_k<64,64,3,3,1,1, 32,32, 32,32,false><<<4096,256,0,stream>>>(buf1, ew3p,  eb3, nullptr, buf2, 1);
    wmma_conv_k<64,64,3,3,1,1, 32,32, 32,32,false><<<4096,256,0,stream>>>(buf2, rew1p, nullptr, nullptr, buf3, 2);
    wmma_conv_k<64,64,1,1,1,0, 32,32, 32,32,false><<<4096,256,0,stream>>>(buf3, rew2p, nullptr, buf2, buf1, 1);
    wmma_conv_k<64,64,1,1,1,0, 32,32, 32,32,false><<<4096,256,0,stream>>>(buf1, ew4p,  eb4, nullptr, buf2, 1);

    // ---- vector quantization (buf2 = latents) ----
    vq_argmin_k<<<1024, 256, 0, stream>>>(buf2, cbp, cn, inds);
    vq_gather_k<<<32768, 256, 0, stream>>>(buf2, cb, inds, buf1, bsum);
    vq_loss_reduce_k<<<1, 256, 0, stream>>>(bsum, 32768, losses);

    // ---- decoder (buf1 = quantized latents) ----
    wmma_conv_k<64,64,3,3,1,1, 32,32, 32,32,false><<<4096,256,0,stream>>>(buf1, dw1p,  db1, nullptr, buf3, 1);
    wmma_conv_k<64,64,3,3,1,1, 32,32, 32,32,false><<<4096,256,0,stream>>>(buf3, rdw1p, nullptr, nullptr, buf2, 2);
    wmma_conv_k<64,64,1,1,1,0, 32,32, 32,32,false><<<4096,256,0,stream>>>(buf2, rdw2p, nullptr, buf3, buf1, 1);
    // fused upsample(32->64) + conv 64->32 + lrelu
    wmma_conv_k<64,32,3,3,1,1, 64,64, 64,64,true><<<8192,256,0,stream>>>(buf1, dw2p, db2, nullptr, buf0, 1);
    // fused upsample(64->128) + conv 32->1 + relu
    dec3_k<<<8192, 256, 0, stream>>>(buf0, dw3, db3, recon);
}
